// MyRNN_69080253989177
// MI455X (gfx1250) — matrix operand order repaired and checked by reading, not compiled, not run on hardware
//
#include <hip/hip_runtime.h>
#include <hip/hip_bf16.h>

// ---------- types ----------
typedef __attribute__((ext_vector_type(16))) __bf16 v16bf;
typedef __attribute__((ext_vector_type(8)))  float  v8f;

union Frag16 { v16bf v; uint4 q[2]; };

// ---------- bf16 helpers (bit-exact RNE, no reliance on __bf16 arithmetic) ----------
__device__ __forceinline__ unsigned short f2bf(float x) {
    unsigned u = __float_as_uint(x);
    unsigned r = u + 0x7FFFu + ((u >> 16) & 1u);
    return (unsigned short)(r >> 16);
}

// ---------- WMMA fragment loads ----------
// A fragment (16x32 bf16, row-major A with leading dim lda).
// lane 0-15: row M=lane,   halves 0-7 -> K=k0+0..7,  halves 8-15 -> K=k0+16..23
// lane16-31: row M=lane-16,halves 0-7 -> K=k0+8..15, halves 8-15 -> K=k0+24..31
__device__ __forceinline__ v16bf load_a_frag(const unsigned short* A, int lda, int k0, int lane) {
    int row  = lane & 15;
    int koff = (lane >> 4) << 3;            // 0 or 8
    const unsigned short* p = A + (size_t)row * lda + k0 + koff;
    Frag16 f;
    f.q[0] = *(const uint4*)(p);            // halves 0-7
    f.q[1] = *(const uint4*)(p + 16);       // halves 8-15
    return f.v;
}

// B fragment (32x16 bf16) from transposed weights Wt[N][K] (leading dim ldk).
// lane: N = n0 + (lane&15); lanes 0-15 hold K=k0+0..7,k0+16..23, lanes 16-31 K=k0+8..15,k0+24..31
// (same K order as the A fragment) -> two 16-byte loads per lane.
__device__ __forceinline__ v16bf load_b_frag(const unsigned short* Wt, int ldk, int n0, int k0, int lane) {
    int n  = n0 + (lane & 15);
    int kb = (lane >> 4) << 3;              // 0 or 8
    const unsigned short* p = Wt + (size_t)n * ldk + k0 + kb;
    Frag16 f;
    f.q[0] = *(const uint4*)(p);
    f.q[1] = *(const uint4*)(p + 16);
    return f.v;
}

__device__ __forceinline__ v8f wmma_bf16(v16bf a, v16bf b, v8f c) {
    return __builtin_amdgcn_wmma_f32_16x16x32_bf16(false, a, false, b, (short)0, c, false, false);
}

static __device__ __forceinline__ v8f vzero8() {
    v8f z = {0.f,0.f,0.f,0.f,0.f,0.f,0.f,0.f};
    return z;
}

// ---------- copy-free 2-stage pipelined wave GEMM tile: 16 x (16*NT), K fixed ----------
template<int NT, int K>
__device__ __forceinline__ void gemm_tile(const unsigned short* __restrict__ A, int lda,
                                          const unsigned short* __restrict__ Wt, int ldk,
                                          int col0, int lane, v8f acc[NT]) {
    v16bf a0 = load_a_frag(A, lda, 0, lane);
    v16bf b0[NT];
    #pragma unroll
    for (int j = 0; j < NT; ++j) b0[j] = load_b_frag(Wt, ldk, col0 + 16 * j, 0, lane);

    #pragma unroll 1
    for (int k0 = 0; k0 < K - 64; k0 += 64) {
        v16bf a1 = load_a_frag(A, lda, k0 + 32, lane);
        v16bf b1[NT];
        #pragma unroll
        for (int j = 0; j < NT; ++j) b1[j] = load_b_frag(Wt, ldk, col0 + 16 * j, k0 + 32, lane);
        #pragma unroll
        for (int j = 0; j < NT; ++j) acc[j] = wmma_bf16(a0, b0[j], acc[j]);

        a0 = load_a_frag(A, lda, k0 + 64, lane);
        #pragma unroll
        for (int j = 0; j < NT; ++j) b0[j] = load_b_frag(Wt, ldk, col0 + 16 * j, k0 + 64, lane);
        #pragma unroll
        for (int j = 0; j < NT; ++j) acc[j] = wmma_bf16(a1, b1[j], acc[j]);
    }

    // tail: k0 = K-64 (peeled; no conditional inside the loop)
    {
        v16bf a1 = load_a_frag(A, lda, K - 32, lane);
        v16bf b1[NT];
        #pragma unroll
        for (int j = 0; j < NT; ++j) b1[j] = load_b_frag(Wt, ldk, col0 + 16 * j, K - 32, lane);
        #pragma unroll
        for (int j = 0; j < NT; ++j) acc[j] = wmma_bf16(a0, b0[j], acc[j]);
        #pragma unroll
        for (int j = 0; j < NT; ++j) acc[j] = wmma_bf16(a1, b1[j], acc[j]);
    }
}

// ---------- prep kernels ----------
__global__ void k_f2bf(const float* __restrict__ in, unsigned short* __restrict__ out, int n) {
    int i = blockIdx.x * blockDim.x + threadIdx.x;
    if (i < n) out[i] = f2bf(in[i]);
}

// in: f32 [R,C] row-major  ->  out: bf16 [C,R] (transposed)
__global__ void k_transpose_bf(const float* __restrict__ in, unsigned short* __restrict__ out,
                               int R, int C) {
    int i = blockIdx.x * blockDim.x + threadIdx.x;
    if (i >= R * C) return;
    int r = i / C, c = i % C;
    out[(size_t)c * R + r] = f2bf(in[i]);
}

// ---------- kernel 1: xW[M=B*T, H=512] = x_bf16[M, I=256] @ Wxh_t + b_h ----------
__global__ __launch_bounds__(256) void k_gemm_xw(const unsigned short* __restrict__ xb,
                                                 const unsigned short* __restrict__ Wt,
                                                 const float* __restrict__ bh,
                                                 float* __restrict__ xW) {
    const int I = 256, Hd = 512;
    int lane = threadIdx.x & 31;
    int wave = threadIdx.x >> 5;
    int row0 = blockIdx.x * 128 + wave * 16;
    int col0 = blockIdx.y * 64;
    const unsigned short* A = xb + (size_t)row0 * I;

    v8f acc[4];
    #pragma unroll
    for (int j = 0; j < 4; ++j) acc[j] = vzero8();

    gemm_tile<4, 256>(A, I, Wt, I, col0, lane, acc);

    int mbase = (lane >> 4) * 8;
    int nl = lane & 15;
    #pragma unroll
    for (int j = 0; j < 4; ++j) {
        int c = col0 + 16 * j + nl;
        float bias = bh[c];
        #pragma unroll
        for (int v = 0; v < 8; ++v) {
            int r = row0 + mbase + v;
            xW[(size_t)r * Hd + c] = acc[j][v] + bias;
        }
    }
}

// ---------- kernel 2 (per time step): Hnext = tanh(xW[:,t,:] + Hprev @ Whh_t) ----------
// Block b covers all 64 batch rows x cols [128b, 128b+128). The xW additive tile
// (64 x 128 f32 = 32KB) is staged global->LDS with CDNA5 async-to-LDS copies
// (ASYNCcnt) issued at kernel entry; the copies overlap the whole WMMA K-loop,
// then s_wait_asynccnt + barrier before the epilogue reads them from LDS.
__global__ __launch_bounds__(256) void k_step(const unsigned short* __restrict__ Hprev,
                                              const unsigned short* __restrict__ Whh_t,
                                              const float* __restrict__ xW,
                                              int t,
                                              unsigned short* __restrict__ Hnext,
                                              float* __restrict__ hT /*may be null*/) {
    const int T = 1024, Hd = 512;
    __shared__ float xw_tile[64 * 128];

    int tid = threadIdx.x;
    int lane = tid & 31;
    int colbase = blockIdx.x * 128;

    // --- issue async global->LDS copies of the xW tile (one 512B row per wave op) ---
    {
        int f  = lane * 4;          // float4 column within the 128-wide row
        int r0 = tid >> 5;          // wave index: rows r0, r0+8, ..., r0+56
        #pragma unroll
        for (int p = 0; p < 8; ++p) {
            int r = r0 + p * 8;     // batch row
            const float* g = xW + ((size_t)r * T + t) * Hd + colbase + f;
            unsigned lds_off = (unsigned)(size_t)&xw_tile[r * 128 + f];
            unsigned long long ga = (unsigned long long)(size_t)g;
            asm volatile("global_load_async_to_lds_b128 %0, %1, off"
                         :: "v"(lds_off), "v"(ga) : "memory");
        }
    }

    int gw = blockIdx.x * 8 + (tid >> 5);
    int row0 = (gw & 3) * 16;
    int col0 = (gw >> 2) * 64;
    const unsigned short* A = Hprev + (size_t)row0 * Hd;

    v8f acc[4];
    #pragma unroll
    for (int j = 0; j < 4; ++j) acc[j] = vzero8();

    gemm_tile<4, 512>(A, Hd, Whh_t, Hd, col0, lane, acc);

    // async copies done + visible to all waves in the workgroup
    asm volatile("s_wait_asynccnt 0x0" ::: "memory");
    __syncthreads();

    int mbase = (lane >> 4) * 8;
    int nl = lane & 15;
    #pragma unroll
    for (int j = 0; j < 4; ++j) {
        int c = col0 + 16 * j + nl;
        int cc = c - colbase;                             // column within LDS tile
        #pragma unroll
        for (int v = 0; v < 8; ++v) {
            int bi = row0 + mbase + v;                    // batch index
            float z = acc[j][v] + xw_tile[bi * 128 + cc];
            float hv = tanhf(z);
            Hnext[(size_t)bi * Hd + c] = f2bf(hv);
            if (hT) hT[(size_t)bi * Hd + c] = hv;
        }
    }
}

// ---------- kernel 3: out[b,t,:] = softmax(Hall[t+1][b] @ Why_t + b_y) ----------
__global__ __launch_bounds__(256) void k_out(const unsigned short* __restrict__ Hall,
                                             const unsigned short* __restrict__ Why_t,
                                             const float* __restrict__ by,
                                             float* __restrict__ out) {
    const int B = 64, T = 1024, Hd = 512, O = 256;
    __shared__ float lds[16][260];
    int lane = threadIdx.x & 31;
    int wave = threadIdx.x >> 5;
    int t = blockIdx.x;
    int b0 = blockIdx.y * 16;
    const unsigned short* A = Hall + ((size_t)(t + 1) * B + b0) * Hd;
    int col0 = wave * 32;

    v8f acc[2];
    acc[0] = vzero8(); acc[1] = vzero8();

    gemm_tile<2, 512>(A, Hd, Why_t, Hd, col0, lane, acc);

    int mbase = (lane >> 4) * 8;
    int nl = lane & 15;
    #pragma unroll
    for (int j = 0; j < 2; ++j) {
        int c = col0 + 16 * j + nl;
        float bias = by[c];
        #pragma unroll
        for (int v = 0; v < 8; ++v)
            lds[mbase + v][c] = acc[j][v] + bias;
    }
    __syncthreads();

    // softmax: wave w handles rows 2w, 2w+1; lane covers cols lane+32*i
    #pragma unroll
    for (int rr = 0; rr < 2; ++rr) {
        int r = wave * 2 + rr;
        float vals[8];
        float m = -3.4e38f;
        #pragma unroll
        for (int i = 0; i < 8; ++i) { vals[i] = lds[r][lane + 32 * i]; m = fmaxf(m, vals[i]); }
        #pragma unroll
        for (int off = 16; off > 0; off >>= 1) m = fmaxf(m, __shfl_xor(m, off, 32));
        float s = 0.f;
        #pragma unroll
        for (int i = 0; i < 8; ++i) { vals[i] = expf(vals[i] - m); s += vals[i]; }
        #pragma unroll
        for (int off = 16; off > 0; off >>= 1) s += __shfl_xor(s, off, 32);
        float inv = 1.f / s;
        size_t base = ((size_t)(b0 + r) * T + t) * O;
        #pragma unroll
        for (int i = 0; i < 8; ++i) out[base + lane + 32 * i] = vals[i] * inv;
    }
}

// ---------- host ----------
extern "C" void kernel_launch(void* const* d_in, const int* in_sizes, int n_in,
                              void* d_out, int out_size, void* d_ws, size_t ws_size,
                              hipStream_t stream) {
    const int B = 64, T = 1024, I = 256, H = 512, O = 256;
    const float* x    = (const float*)d_in[0];
    const float* h0   = (const float*)d_in[1];
    const float* Wxh  = (const float*)d_in[2];
    const float* Whh  = (const float*)d_in[3];
    const float* bh   = (const float*)d_in[4];
    const float* Why  = (const float*)d_in[5];
    const float* byv  = (const float*)d_in[6];
    float* out = (float*)d_out;

    // workspace carve-up (all offsets 256B-aligned by construction)
    char* w = (char*)d_ws;
    size_t off = 0;
    float*          xW    = (float*)(w + off);          off += (size_t)B * T * H * 4;
    unsigned short* xb    = (unsigned short*)(w + off); off += (size_t)B * T * I * 2;
    unsigned short* Wxh_t = (unsigned short*)(w + off); off += (size_t)H * I * 2;
    unsigned short* Whh_t = (unsigned short*)(w + off); off += (size_t)H * H * 2;
    unsigned short* Why_t = (unsigned short*)(w + off); off += (size_t)O * H * 2;
    unsigned short* Hall  = (unsigned short*)(w + off); off += (size_t)(T + 1) * B * H * 2;
    (void)ws_size; (void)in_sizes; (void)n_in; (void)out_size;

    // prep: bf16 conversions + weight transposes
    {
        int n = B * T * I;
        k_f2bf<<<(n + 255) / 256, 256, 0, stream>>>(x, xb, n);
        n = B * H;
        k_f2bf<<<(n + 255) / 256, 256, 0, stream>>>(h0, Hall, n);      // slot 0 = h_0
        n = I * H;
        k_transpose_bf<<<(n + 255) / 256, 256, 0, stream>>>(Wxh, Wxh_t, I, H);
        n = H * H;
        k_transpose_bf<<<(n + 255) / 256, 256, 0, stream>>>(Whh, Whh_t, H, H);
        n = H * O;
        k_transpose_bf<<<(n + 255) / 256, 256, 0, stream>>>(Why, Why_t, H, O);
    }

    // kernel 1: xW = x @ W_xh + b_h   (M=B*T=65536 rows)
    {
        dim3 grid((B * T) / 128, H / 64);
        k_gemm_xw<<<grid, 256, 0, stream>>>(xb, Wxh_t, bh, xW);
    }

    // kernel 2: sequential recurrence (1024 tiny launches on the critical path)
    float* hT = out + (size_t)B * T * O;
    for (int t = 0; t < T; ++t) {
        const unsigned short* Hprev = Hall + (size_t)t * B * H;
        unsigned short*       Hnext = Hall + (size_t)(t + 1) * B * H;
        k_step<<<4, 256, 0, stream>>>(Hprev, Whh_t, xW, t, Hnext,
                                      (t == T - 1) ? hT : (float*)nullptr);
    }

    // kernel 3: output projection + fused softmax, fully parallel over (t, b-slab)
    {
        dim3 grid(T, B / 16);
        k_out<<<grid, 256, 0, stream>>>(Hall, Why_t, byv, out);
    }
}
